// CausalSelfAttention_10900626997556
// MI455X (gfx1250) — matrix operand order repaired and checked by reading, not compiled, not run on hardware
//
#include <hip/hip_runtime.h>
#include <hip/hip_bf16.h>

// ---------------------------------------------------------------------------
// CDNA5 (gfx1250) causal self-attention, bf16 WMMA path + async-LDS GEMM.
//   D_MODEL=2048, N_HEADS=16, HEAD_DIM=128, B=4, S=2048
// d_out = [out (B,S,D) f32][k (B,H,S,Dh) f32][v (B,H,S,Dh) f32]
// ---------------------------------------------------------------------------

typedef __attribute__((ext_vector_type(16))) __bf16 v16bf;
typedef __attribute__((ext_vector_type(8)))  __bf16 v8bf;
typedef __attribute__((ext_vector_type(8)))  float  v8f;

union BFU { v16bf v; v8bf h[2]; };

__device__ __forceinline__ v16bf frag2(const __bf16* p0, const __bf16* p1) {
    BFU u;
    u.h[0] = *reinterpret_cast<const v8bf*>(p0);
    u.h[1] = *reinterpret_cast<const v8bf*>(p1);
    return u.v;
}

__device__ __forceinline__ v8f wmma_bf16(v16bf a, v16bf b, v8f c) {
    return __builtin_amdgcn_wmma_f32_16x16x32_bf16(
        /*neg_a=*/false, a, /*neg_b=*/false, b,
        /*c_mod=*/(short)0, c, /*reuse_a=*/false, /*reuse_b=*/false);
}

__device__ __forceinline__ v8f zero8() {
    v8f z;
    #pragma unroll
    for (int i = 0; i < 8; ++i) z[i] = 0.0f;
    return z;
}

// Generic-pointer low 32 bits of a __shared__ object == wave-relative LDS
// byte offset (flat aperture mapping, ISA 10.2).
__device__ __forceinline__ unsigned lds_off(const void* p) {
    return (unsigned)(unsigned long long)p;
}

__device__ __forceinline__ void async_b128(unsigned lds_byte, const void* gaddr) {
    asm volatile("global_load_async_to_lds_b128 %0, %1, off"
                 :: "v"(lds_byte), "v"((unsigned long long)gaddr) : "memory");
}

// ---------------------------------------------------------------------------
// f32 -> bf16 bulk conversion (one-time; everything then lives in L2 as bf16)
// ---------------------------------------------------------------------------
__global__ __launch_bounds__(256) void to_bf16(const float* __restrict__ src,
                                               __bf16* __restrict__ dst, int n) {
    const int i = (blockIdx.x * 256 + threadIdx.x) * 4;
    if (i < n) {
        const float4 v = *reinterpret_cast<const float4*>(src + i);
        dst[i]     = (__bf16)v.x;
        dst[i + 1] = (__bf16)v.y;
        dst[i + 2] = (__bf16)v.z;
        dst[i + 3] = (__bf16)v.w;
    }
}

// ---------------------------------------------------------------------------
// NT GEMM:  C[M,N] = A[M,K] * W[N,K]^T, both operands bf16.
// Block: 256 thr = 8 waves; block tile 128x128; wave tile 64(M)x32(N).
// K-loop double-buffered via global_load_async_to_lds_b128 (ASYNCcnt).
// ---------------------------------------------------------------------------
__global__ __launch_bounds__(256) void gemm_nt(
    const __bf16* __restrict__ A, const __bf16* __restrict__ W,
    float* __restrict__ C, int M, int N, int K)
{
    constexpr int LDST = 40;                  // 32 + 8 pad (80 B rows, 16B aligned)
    __shared__ __bf16 ldsA[2][128 * LDST];
    __shared__ __bf16 ldsB[2][128 * LDST];

    const int tid  = threadIdx.x;
    const int lane = tid & 31;
    const int wave = tid >> 5;
    const int wm   = wave >> 2;               // 0..1 : M half (64 rows)
    const int wn   = wave & 3;                // 0..3 : N quarter (32 cols)
    const int blkM = blockIdx.y * 128;
    const int blkN = blockIdx.x * 128;
    const int lrow = lane & 15;
    const bool hi  = lane >= 16;

    const unsigned aBase = lds_off(&ldsA[0][0]);
    const unsigned bBase = lds_off(&ldsB[0][0]);

    // Stage one 128x32 bf16 tile per operand: 512 chunks of 16B, 2 per thread.
    auto issue_tile = [&](int buf, int kt) {
        #pragma unroll
        for (int p = 0; p < 2; ++p) {
            const int c   = p * 256 + tid;
            const int row = c >> 2;
            const int co  = (c & 3) * 8;
            const unsigned lby = (unsigned)((buf * 128 * LDST + row * LDST + co) * 2);
            async_b128(aBase + lby, A + (size_t)(blkM + row) * K + kt + co);
            async_b128(bBase + lby, W + (size_t)(blkN + row) * K + kt + co);
        }
    };

    v8f acc[4][2];
    #pragma unroll
    for (int i = 0; i < 4; ++i)
        #pragma unroll
        for (int j = 0; j < 2; ++j) acc[i][j] = zero8();

    const int NT = K / 32;
    issue_tile(0, 0);
    for (int n = 0; n < NT; ++n) {
        const int cur = n & 1;
        if (n + 1 < NT) {
            issue_tile(cur ^ 1, (n + 1) * 32);     // prefetch next tile
            asm volatile("s_wait_asynccnt 4" ::: "memory");  // tile n landed
        } else {
            asm volatile("s_wait_asynccnt 0" ::: "memory");
        }
        __syncthreads();

        const __bf16* la = &ldsA[cur][0];
        const __bf16* lb = &ldsB[cur][0];
        const int aklo = hi ? 8 : 0;          // A: K 0..7/16..23 vs 8..15/24..31
        v16bf afr[4];
        #pragma unroll
        for (int i = 0; i < 4; ++i) {
            const int r = wm * 64 + i * 16 + lrow;
            afr[i] = frag2(&la[r * LDST + aklo], &la[r * LDST + 16 + aklo]);
        }
        const int bk = hi ? 8 : 0;            // B: K 0..7/16..23 vs 8..15/24..31
        v16bf bfr[2];
        #pragma unroll
        for (int j = 0; j < 2; ++j) {
            const int r = wn * 32 + j * 16 + lrow;
            bfr[j] = frag2(&lb[r * LDST + bk], &lb[r * LDST + bk + 16]);
        }
        #pragma unroll
        for (int i = 0; i < 4; ++i)
            #pragma unroll
            for (int j = 0; j < 2; ++j)
                acc[i][j] = wmma_bf16(afr[i], bfr[j], acc[i][j]);

        asm volatile("s_wait_dscnt 0" ::: "memory");  // reads drained before
        __syncthreads();                              // buffer is overwritten
    }

    // ---- store C: element (m = r + 8*hi, n = lane&15) ----
    #pragma unroll
    for (int i = 0; i < 4; ++i) {
        #pragma unroll
        for (int j = 0; j < 2; ++j) {
            const int gm = blkM + wm * 64 + i * 16 + (hi ? 8 : 0);
            const int gn = blkN + wn * 32 + j * 16 + lrow;
            #pragma unroll
            for (int r = 0; r < 8; ++r)
                C[(size_t)(gm + r) * N + gn] = acc[i][j][r];
        }
    }
}

// ---------------------------------------------------------------------------
// RoPE + layout prep. One block per (b,s) token row, 256 threads.
//  - Qbf, Kbf : bf16 (b,h,s,d) post-rope
//  - Kout,Vout: f32 (b,h,s,d) straight into d_out return slots
//  - Vt       : bf16 (b,h,d,s) transposed V for the P*V WMMA B-fragment
// ---------------------------------------------------------------------------
__global__ __launch_bounds__(256) void rope_prep(
    const float* __restrict__ Y, const float* __restrict__ cosT,
    const float* __restrict__ sinT,
    float* __restrict__ Kout, float* __restrict__ Vout,
    __bf16* __restrict__ Qbf, __bf16* __restrict__ Kbf, __bf16* __restrict__ Vt)
{
    const int m = blockIdx.x;                 // 0..8191
    const int b = m >> 11, s = m & 2047;
    const int p = threadIdx.x & 63;           // rotary pair index
    const int h0 = threadIdx.x >> 6;          // 4 heads per thread

    const float c  = cosT[s * 64 + p];
    const float sn = sinT[s * 64 + p];
    const float* yrow = Y + (size_t)m * 6144;

    for (int h = h0; h < 16; h += 4) {
        const size_t hd = ((size_t)(b * 16 + h) * 2048 + s) * 128 + 2 * p;
        // Q (columns 0..2047)
        float x1 = yrow[h * 128 + 2 * p];
        float x2 = yrow[h * 128 + 2 * p + 1];
        Qbf[hd]     = (__bf16)(x1 * c - x2 * sn);
        Qbf[hd + 1] = (__bf16)(x1 * sn + x2 * c);
        // K (columns 2048..4095) -- post-rope K is a returned output
        x1 = yrow[2048 + h * 128 + 2 * p];
        x2 = yrow[2048 + h * 128 + 2 * p + 1];
        const float k0 = x1 * c - x2 * sn;
        const float k1 = x1 * sn + x2 * c;
        Kout[hd] = k0;  Kout[hd + 1] = k1;
        Kbf[hd] = (__bf16)k0;  Kbf[hd + 1] = (__bf16)k1;
        // V (columns 4096..6143)
        const float v0 = yrow[4096 + h * 128 + 2 * p];
        const float v1 = yrow[4096 + h * 128 + 2 * p + 1];
        Vout[hd] = v0;  Vout[hd + 1] = v1;
        const size_t vt = ((size_t)(b * 16 + h) * 128 + 2 * p) * 2048 + s;
        Vt[vt]        = (__bf16)v0;
        Vt[vt + 2048] = (__bf16)v1;
    }
}

// ---------------------------------------------------------------------------
// Flash attention. Grid = B*H*(S/128) blocks, 8 waves; each wave owns 16
// query rows, streams causal key tiles of 32, online softmax in f32.
// P (16x32) goes C-layout -> LDS (bf16) -> A-fragment layout.
// Output written bf16 (b,s,h*d) for the final projection GEMM.
// ---------------------------------------------------------------------------
__global__ __launch_bounds__(256) void attn(
    const __bf16* __restrict__ Qbf, const __bf16* __restrict__ Kbf,
    const __bf16* __restrict__ Vt, __bf16* __restrict__ Obf)
{
    __shared__ __bf16 ldsP[8][16 * 40];       // per-wave 16x32 P tile, padded

    const int lane = threadIdx.x & 31;
    const int wave = threadIdx.x >> 5;
    const int bh   = blockIdx.x >> 4;         // 0..63  (B*H)
    const int qblk = blockIdx.x & 15;
    const int q0   = qblk * 128 + wave * 16;
    const int b    = bh >> 4, h = bh & 15;
    const int lrow = lane & 15;
    const bool hi  = lane >= 16;
    const float scale = 0.08838834764831845f; // 1/sqrt(128)

    const size_t base = (size_t)bh * 2048 * 128;
    const __bf16* Kb  = Kbf + base;
    const __bf16* Vb  = Vt + base;            // (d, s) layout, same extent
    __bf16* myP = &ldsP[wave][0];

    // Q fragments: 16 rows x 128 (4 K-steps of 32)
    v16bf qf[4];
    {
        const __bf16* qrow = Qbf + base + (size_t)(q0 + lrow) * 128;
        const int klo = hi ? 8 : 0;
        #pragma unroll
        for (int kk = 0; kk < 4; ++kk)
            qf[kk] = frag2(qrow + kk * 32 + klo, qrow + kk * 32 + 16 + klo);
    }

    v8f o[8];
    #pragma unroll
    for (int d = 0; d < 8; ++d) o[d] = zero8();
    float mrun[8], lrun[8];
    #pragma unroll
    for (int r = 0; r < 8; ++r) { mrun[r] = -3.0e38f; lrun[r] = 0.0f; }

    const int qmax = q0 + 15;
    for (int j = 0; j <= qmax; j += 32) {
        // ---- scores: S = Q * K^T, 16 x 32 (two 16-wide accumulators) ----
        v8f s0 = zero8(), s1 = zero8();
        const int kd = hi ? 8 : 0;
        #pragma unroll
        for (int kk = 0; kk < 4; ++kk) {
            const __bf16* k0p = Kb + (size_t)(j + lrow) * 128 + kk * 32 + kd;
            const v16bf kf0 = frag2(k0p, k0p + 16);
            const __bf16* k1p = k0p + 16 * 128;
            const v16bf kf1 = frag2(k1p, k1p + 16);
            s0 = wmma_bf16(qf[kk], kf0, s0);
            s1 = wmma_bf16(qf[kk], kf1, s1);
        }
        // ---- online softmax, row groups = 16-lane halves ----
        float corr[8];
        const int rowb = q0 + (hi ? 8 : 0);
        #pragma unroll
        for (int r = 0; r < 8; ++r) {
            const int qrow = rowb + r;
            const int key0 = j + lrow, key1 = key0 + 16;
            float v0 = (key0 <= qrow) ? s0[r] * scale : -3.0e38f;
            float v1 = (key1 <= qrow) ? s1[r] * scale : -3.0e38f;
            float mx = fmaxf(v0, v1);
            #pragma unroll
            for (int off = 8; off >= 1; off >>= 1)
                mx = fmaxf(mx, __shfl_xor(mx, off, 32));
            const float nm = fmaxf(mrun[r], mx);
            const float p0 = __expf(v0 - nm);
            const float p1 = __expf(v1 - nm);
            const float cf = __expf(mrun[r] - nm);
            float ps = p0 + p1;
            #pragma unroll
            for (int off = 8; off >= 1; off >>= 1)
                ps += __shfl_xor(ps, off, 32);
            lrun[r] = lrun[r] * cf + ps;
            mrun[r] = nm;
            corr[r] = cf;
            const int ml = r + (hi ? 8 : 0);
            myP[ml * 40 + lrow]      = (__bf16)p0;
            myP[ml * 40 + lrow + 16] = (__bf16)p1;
        }
        #pragma unroll
        for (int d = 0; d < 8; ++d)
            #pragma unroll
            for (int r = 0; r < 8; ++r) o[d][r] *= corr[r];

        asm volatile("s_wait_dscnt 0" ::: "memory");   // P in LDS visible

        // ---- P fragment (A layout) ----
        const int klo = hi ? 8 : 0;
        const v16bf pf = frag2(&myP[lrow * 40 + klo], &myP[lrow * 40 + 16 + klo]);

        // ---- O += P * V  (B-frag from V^T: lane = d column, K 0..7/16..23 vs 8..15/24..31) ----
        #pragma unroll
        for (int d = 0; d < 8; ++d) {
            const __bf16* vp = Vb + (size_t)(d * 16 + lrow) * 2048 + j + (hi ? 8 : 0);
            const v16bf vf = frag2(vp, vp + 16);
            o[d] = wmma_bf16(pf, vf, o[d]);
        }
    }

    // ---- normalize + store bf16 O in (b, s, h*128 + d) layout ----
    float inv[8];
    #pragma unroll
    for (int r = 0; r < 8; ++r) inv[r] = 1.0f / lrun[r];
    const size_t orow0 = ((size_t)b * 2048 + q0 + (hi ? 8 : 0)) * 2048 + h * 128;
    #pragma unroll
    for (int d = 0; d < 8; ++d)
        #pragma unroll
        for (int r = 0; r < 8; ++r)
            Obf[orow0 + (size_t)r * 2048 + d * 16 + lrow] = (__bf16)(o[d][r] * inv[r]);
}

// ---------------------------------------------------------------------------
extern "C" void kernel_launch(void* const* d_in, const int* in_sizes, int n_in,
                              void* d_out, int out_size, void* d_ws, size_t ws_size,
                              hipStream_t stream) {
    (void)in_sizes; (void)n_in; (void)out_size; (void)ws_size;
    const float* x     = (const float*)d_in[0];   // (4,2048,2048)
    const float* rcos  = (const float*)d_in[1];   // (2048,64)
    const float* rsin  = (const float*)d_in[2];   // (2048,64)
    const float* qkv_w = (const float*)d_in[3];   // (6144,2048)
    const float* out_w = (const float*)d_in[4];   // (2048,2048)

    float* out  = (float*)d_out;                  // (4,2048,2048)
    float* kout = out + (size_t)16777216;         // (4,16,2048,128)
    float* vout = kout + (size_t)16777216;        // (4,16,2048,128)

    char* wsb = (char*)d_ws;
    float*  Yraw   = (float*)wsb;                                  // 192 MiB
    __bf16* Qbf    = (__bf16*)(wsb + 201326592ull);                // 32 MiB
    __bf16* Kbf    = (__bf16*)(wsb + 234881024ull);                // 32 MiB
    __bf16* Vt     = (__bf16*)(wsb + 268435456ull);                // 32 MiB
    __bf16* Obf    = (__bf16*)(wsb + 301989888ull);                // 32 MiB
    __bf16* Xbf    = (__bf16*)(wsb + 335544320ull);                // 32 MiB
    __bf16* Wqkvbf = (__bf16*)(wsb + 369098752ull);                // 24 MiB
    __bf16* Woutbf = (__bf16*)(wsb + 394264576ull);                //  8 MiB

    // 0) one-time f32 -> bf16 conversions (all live in the 192MB L2 after this)
    to_bf16<<<16777216 / 1024, 256, 0, stream>>>(x, Xbf, 16777216);
    to_bf16<<<12582912 / 1024, 256, 0, stream>>>(qkv_w, Wqkvbf, 12582912);
    to_bf16<<<4194304 / 1024, 256, 0, stream>>>(out_w, Woutbf, 4194304);
    // 1) QKV projection: Y = X * Wqkv^T   (8192 x 6144 x 2048)
    gemm_nt<<<dim3(6144 / 128, 8192 / 128), 256, 0, stream>>>(
        Xbf, Wqkvbf, Yraw, 8192, 6144, 2048);
    // 2) RoPE + K/V emission + bf16/transposed staging
    rope_prep<<<8192, 256, 0, stream>>>(Yraw, rcos, rsin, kout, vout, Qbf, Kbf, Vt);
    // 3) Causal flash attention
    attn<<<4 * 16 * (2048 / 128), 256, 0, stream>>>(Qbf, Kbf, Vt, Obf);
    // 4) Output projection: out = O * Wout^T  (8192 x 2048 x 2048)
    gemm_nt<<<dim3(2048 / 128, 8192 / 128), 256, 0, stream>>>(
        Obf, Woutbf, out, 8192, 2048, 2048);
}
